// AGCRN_82703890252046
// MI455X (gfx1250) — matrix operand order repaired and checked by reading, not compiled, not run on hardware
//
#include <hip/hip_runtime.h>
#include <hip/hip_bf16.h>
#include <cstdint>
#include <math.h>

// ---------------------------------------------------------------------------
// AGCRN on gfx1250 (MI455X): WMMA f16->f32 pipeline, direct-global fragments,
// 2x2 register tiling in the graph GEMM, TDM staging for layer-1 node tiles.
//   B=64, T=12, N=883, DIN=1, HID=64, EMB=10, K=2, HOR=12, ODIM=1
// ---------------------------------------------------------------------------

typedef __attribute__((ext_vector_type(16))) _Float16 v16h;
typedef __attribute__((ext_vector_type(8)))  _Float16 v8h;
typedef __attribute__((ext_vector_type(8)))  float    v8f;
typedef __attribute__((ext_vector_type(4)))  unsigned int v4u;
typedef __attribute__((ext_vector_type(8)))  int      v8i;
typedef __attribute__((ext_vector_type(4)))  int      v4i;
typedef _Float16 f16;

#define NNODE 883
#define NPAD  896      // 56*16 == 28*32
#define BATCH 64
#define TSTEPS 12
#define HIDC 64
#define EMBD 10
#define CIN0 65        // DIN + HID
#define CIN1 128       // HID + HID
#define K2P0 160       // 2*CIN0 padded to mult of 32
#define K2P1 256       // 2*CIN1
#define NC0 (BATCH * CIN0)   // 4160 (mult of 32)
#define NC1 (BATCH * CIN1)   // 8192

#ifdef __has_builtin
#if __has_builtin(__builtin_amdgcn_tensor_load_to_lds) && \
    __has_builtin(__builtin_amdgcn_s_wait_tensorcnt)
#define USE_TDM 1
#endif
#endif
#ifndef USE_TDM
#define USE_TDM 0
#endif

// ---------------- TDM 2D tile load: global -> LDS --------------------------
#if USE_TDM
__device__ __forceinline__ void tdm_load_2d(uint32_t lds_addr, const void* gptr,
                                            uint32_t width, uint32_t rows,
                                            uint32_t row_stride_elems,
                                            uint32_t pad_int_code,
                                            uint32_t pad_amt_code) {
  uint64_t ga = (uint64_t)gptr;
  v4u g0;
  g0[0] = 1u;                                                // count=1
  g0[1] = lds_addr;                                          // lds_addr
  g0[2] = (uint32_t)ga;                                      // global_addr lo
  g0[3] = (uint32_t)((ga >> 32) & 0x01FFFFFFu) | (2u << 30); // hi | type=2
  v8i g1;
  g1[0] = (int)((1u << 16) |                  // data_size = 1 (2 bytes)
                (1u << 20) |                  // pad_enable
                (pad_int_code << 22) | (pad_amt_code << 25));
  g1[1] = (int)((width & 0xFFFFu) << 16);                    // tensor_dim0 lo
  g1[2] = (int)(((width >> 16) & 0xFFFFu) | ((rows & 0xFFFFu) << 16));
  g1[3] = (int)(((rows >> 16) & 0xFFFFu) | ((width & 0xFFFFu) << 16)); // tile_dim0
  g1[4] = (int)(rows & 0xFFFFu);                             // tile_dim1, dim2=0
  g1[5] = (int)row_stride_elems;                             // dim0_stride lo
  g1[6] = 0;
  g1[7] = 0;
  v4i gz = {0, 0, 0, 0};
#if __clang_major__ >= 23
  v8i gz8 = {0, 0, 0, 0, 0, 0, 0, 0};
  __builtin_amdgcn_tensor_load_to_lds(g0, g1, gz, gz, gz8, 0);
#else
  __builtin_amdgcn_tensor_load_to_lds(g0, g1, gz, gz, 0);
#endif
}
#endif

// ---------------- fragment helpers -----------------------------------------
__device__ __forceinline__ v16h frag_from_pair(v8h lo, v8h hi) {
  v16h r;
#pragma unroll
  for (int e = 0; e < 8; ++e) { r[e] = lo[e]; r[8 + e] = hi[e]; }
  return r;
}

// ---------------- adjacency: A = softmax(relu(E E^T)), f16, zero-padded ----
__global__ __launch_bounds__(256) void adj_kernel(const float* __restrict__ E,
                                                  f16* __restrict__ A) {
  __shared__ float buf[NNODE];
  __shared__ float red[256];
  __shared__ float en[16];
  int n = blockIdx.x;
  int t = threadIdx.x;
  if (n >= NNODE) {
    for (int m = t; m < NPAD; m += 256) A[(size_t)n * NPAD + m] = (f16)0.f;
    return;
  }
  if (t < EMBD) en[t] = E[n * EMBD + t];
  __syncthreads();
  float lmax = -1e30f;
  for (int m = t; m < NNODE; m += 256) {
    float s = 0.f;
#pragma unroll
    for (int d = 0; d < EMBD; ++d) s += en[d] * E[m * EMBD + d];
    s = fmaxf(s, 0.f);
    buf[m] = s;
    lmax = fmaxf(lmax, s);
  }
  red[t] = lmax;
  __syncthreads();
  for (int o = 128; o > 0; o >>= 1) {
    if (t < o) red[t] = fmaxf(red[t], red[t + o]);
    __syncthreads();
  }
  float mx = red[0];
  __syncthreads();
  float lsum = 0.f;
  for (int m = t; m < NNODE; m += 256) {
    float e = __expf(buf[m] - mx);
    buf[m] = e;
    lsum += e;
  }
  red[t] = lsum;
  __syncthreads();
  for (int o = 128; o > 0; o >>= 1) {
    if (t < o) red[t] += red[t + o];
    __syncthreads();
  }
  float inv = 1.f / red[0];
  for (int m = t; m < NPAD; m += 256)
    A[(size_t)n * NPAD + m] = (f16)((m < NNODE) ? buf[m] * inv : 0.f);
}

// ---------------- weight pooling, transposed + K-padded --------------------
// dst[n][o][p] (p < k2p), p = k*cin + i ; wpool flat [d][k][i][o]
__global__ __launch_bounds__(256) void pool_wT(f16* __restrict__ dst,
                                               const float* __restrict__ wpool,
                                               const float* __restrict__ E,
                                               int cin, int cout, int k2p) {
  size_t idx = (size_t)blockIdx.x * 256 + threadIdx.x;
  size_t total = (size_t)NNODE * cout * k2p;
  if (idx >= total) return;
  int p = (int)(idx % k2p);
  int o = (int)((idx / k2p) % cout);
  int n = (int)(idx / ((size_t)k2p * cout));
  float s = 0.f;
  if (p < 2 * cin) {
    size_t J = (size_t)2 * cin * cout;
#pragma unroll
    for (int d = 0; d < EMBD; ++d)
      s += E[n * EMBD + d] * wpool[(size_t)d * J + (size_t)p * cout + o];
  }
  dst[idx] = (f16)s;
}

__global__ __launch_bounds__(256) void pool_b_kernel(float* __restrict__ dst,
                                                     const float* __restrict__ bpool,
                                                     const float* __restrict__ E, int O) {
  int idx = blockIdx.x * 256 + threadIdx.x;
  if (idx >= NNODE * O) return;
  int n = idx / O, o = idx - n * O;
  float s = 0.f;
#pragma unroll
  for (int d = 0; d < EMBD; ++d) s += E[n * EMBD + d] * bpool[d * O + o];
  dst[idx] = s;
}

__global__ __launch_bounds__(256) void zero_h_kernel(float* __restrict__ h) {
  int idx = blockIdx.x * 256 + threadIdx.x;
  if (idx < NNODE * BATCH * HIDC) h[idx] = 0.f;
}

// ---------------- feature build: dual layout -------------------------------
// Xn[node][b*cin+c] (for node_gemm) and Xt[b*cin+c][node] (for gemm_agg).
__global__ __launch_bounds__(256) void build_x0(f16* __restrict__ Xn, f16* __restrict__ Xt,
                                                const float* __restrict__ src,
                                                const float* __restrict__ h,
                                                const f16* __restrict__ z, int t) {
  int idx = blockIdx.x * 256 + threadIdx.x;
  const int total = NC0 * NNODE;
  if (idx >= total) return;
  int n = idx % NNODE;
  int cg = idx / NNODE;
  int b = cg / CIN0, c = cg % CIN0;
  float v;
  if (c == 0) {
    v = src[((size_t)b * TSTEPS + t) * NNODE + n];
  } else {
    size_t hi = ((size_t)n * BATCH + b) * HIDC + (c - 1);
    v = h[hi];
    if (z) v *= (float)z[hi];
  }
  f16 fv = (f16)v;
  Xt[(size_t)cg * NPAD + n] = fv;
  Xn[(size_t)n * NC0 + cg] = fv;
}

__global__ __launch_bounds__(256) void build_x1(f16* __restrict__ Xn, f16* __restrict__ Xt,
                                                const f16* __restrict__ xseq,
                                                const float* __restrict__ h,
                                                const f16* __restrict__ z) {
  int idx = blockIdx.x * 256 + threadIdx.x;
  const int total = NC1 * NNODE;
  if (idx >= total) return;
  int n = idx % NNODE;
  int cg = idx / NNODE;
  int b = cg / CIN1, c = cg % CIN1;
  f16 fv;
  if (c < HIDC) {
    fv = xseq[((size_t)n * BATCH + b) * HIDC + c];
  } else {
    size_t hi = ((size_t)n * BATCH + b) * HIDC + (c - HIDC);
    float v = h[hi];
    if (z) v *= (float)z[hi];
    fv = (f16)v;
  }
  Xt[(size_t)cg * NPAD + n] = fv;
  Xn[(size_t)n * NC1 + cg] = fv;
}

// ---------------- graph aggregation GEMM: AX = A @ X (LDS-free WMMA) -------
// A: [NPAD][NPAD] f16 zero-padded. Xt: [ncols][NPAD] (feature-major).
// AX out: [NPAD][ncols] (node-major).
// 2x2 register tile per wave: 32 (M) x 32 (N), 4 accumulators sharing frags.
__global__ __launch_bounds__(256) void gemm_agg(const f16* __restrict__ A,
                                                const f16* __restrict__ Xt,
                                                f16* __restrict__ AX, int ncols) {
  int tid = threadIdx.x, wave = tid >> 5, lane = tid & 31;
  int ct = blockIdx.x * 8 + wave;          // 32-col tile index
  int c0 = ct * 32;
  if (c0 >= ncols) return;                 // tail waves (layer0), no barriers
  int m0 = blockIdx.y * 32;
  int mm = lane & 15, hh = lane >> 4;
  const f16* ap0 = A + (size_t)(m0 + mm) * NPAD + hh * 8;
  const f16* ap1 = ap0 + (size_t)16 * NPAD;
  const f16* bp0 = Xt + (size_t)(c0 + mm) * NPAD + hh * 8;
  const f16* bp1 = bp0 + (size_t)16 * NPAD;
  v8f acc00 = {}, acc01 = {}, acc10 = {}, acc11 = {};
  for (int kt = 0; kt < NPAD / 32; ++kt) {
    int k0 = kt * 32;
    v16h af0 = frag_from_pair(*(const v8h*)(ap0 + k0), *(const v8h*)(ap0 + k0 + 16));
    v16h af1 = frag_from_pair(*(const v8h*)(ap1 + k0), *(const v8h*)(ap1 + k0 + 16));
    v16h bf0 = frag_from_pair(*(const v8h*)(bp0 + k0), *(const v8h*)(bp0 + k0 + 16));
    v16h bf1 = frag_from_pair(*(const v8h*)(bp1 + k0), *(const v8h*)(bp1 + k0 + 16));
    acc00 = __builtin_amdgcn_wmma_f32_16x16x32_f16(false, af0, false, bf0, (short)0,
                                                   acc00, false, false);
    acc01 = __builtin_amdgcn_wmma_f32_16x16x32_f16(false, af0, false, bf1, (short)0,
                                                   acc01, false, false);
    acc10 = __builtin_amdgcn_wmma_f32_16x16x32_f16(false, af1, false, bf0, (short)0,
                                                   acc10, false, false);
    acc11 = __builtin_amdgcn_wmma_f32_16x16x32_f16(false, af1, false, bf1, (short)0,
                                                   acc11, false, false);
    if (kt + 1 < NPAD / 32) {                 // -> global_prefetch_b8 (near)
      __builtin_prefetch(ap0 + k0 + 32, 0, 3);
      __builtin_prefetch(ap1 + k0 + 32, 0, 3);
      __builtin_prefetch(bp0 + k0 + 32, 0, 3);
      __builtin_prefetch(bp1 + k0 + 32, 0, 3);
    }
  }
  int mb = m0 + hh * 8;
#pragma unroll
  for (int r = 0; r < 8; ++r) {
    AX[(size_t)(mb + r) * ncols + c0 + mm]           = (f16)acc00[r];
    AX[(size_t)(mb + r) * ncols + c0 + 16 + mm]      = (f16)acc01[r];
    AX[(size_t)(mb + 16 + r) * ncols + c0 + mm]      = (f16)acc10[r];
    AX[(size_t)(mb + 16 + r) * ncols + c0 + 16 + mm] = (f16)acc11[r];
  }
}

// ---------------- per-node GEMM + fused GRU --------------------------------
// node n: out[64, cout] = cat(X_n | AX_n)[64, k2p] @ W_n[k2p, cout] + bias
// W stored transposed [n][cout][k2p] (zero K-padded) -> direct global B-frags.
// The [64][k2p] operand is staged in LDS once (TDM when cin==128).
__global__ __launch_bounds__(256) void node_gemm(const f16* __restrict__ X,
                                                 const f16* __restrict__ AX,
                                                 const f16* __restrict__ W,
                                                 const float* __restrict__ bias,
                                                 int cin, int cout, int k2p, int mode,
                                                 f16* __restrict__ zbuf,
                                                 float* __restrict__ rbuf,
                                                 float* __restrict__ h,
                                                 f16* __restrict__ seq_out) {
  __shared__ __align__(16) f16 Xc[64 * K2P1];
  int n = blockIdx.x;
  int tid = threadIdx.x, wave = tid >> 5, lane = tid & 31;
  int bc = BATCH * cin;
  const f16* Xn = X + (size_t)n * bc;
  const f16* AXn = AX + (size_t)n * bc;
  bool filled = false;
#if USE_TDM
  if (cin == CIN1) {  // rows of 128 f16 = 256B: pad codes interval=5, amount=63
    if (wave == 0) {
      uint32_t lds0 = (uint32_t)(uintptr_t)(&Xc[0]);
      tdm_load_2d(lds0, Xn, 128, 64, 128, 5, 63);
      tdm_load_2d(lds0 + 256, AXn, 128, 64, 128, 5, 63);
      __builtin_amdgcn_s_wait_tensorcnt(0);
    }
    filled = true;
  }
#endif
  if (!filled) {
    int b = tid & 63, g = tid >> 6;
    int per = k2p >> 2;
    int i0 = g * per;
    for (int j = 0; j < per; ++j) {
      int i = i0 + j;
      f16 v = (f16)0.f;
      if (i < cin) v = Xn[b * cin + i];
      else if (i < 2 * cin) v = AXn[b * cin + (i - cin)];
      Xc[b * k2p + i] = v;
    }
  }
  __syncthreads();
  if (wave * 16 >= cout) return;  // no further barriers

  int mm = lane & 15, hh = lane >> 4;
  const f16* wp = W + ((size_t)n * cout + (wave * 16 + mm)) * k2p + hh * 8;
  v8f acc[4] = {};
  int ktiles = k2p / 32;
  for (int kt = 0; kt < ktiles; ++kt) {
    int k0 = kt * 32;
    v8h b0 = *(const v8h*)(wp + k0);
    v8h b1 = *(const v8h*)(wp + k0 + 16);
    v16h bf = frag_from_pair(b0, b1);
#pragma unroll
    for (int mt = 0; mt < 4; ++mt) {
      const f16* xp = Xc + (size_t)(mt * 16 + mm) * k2p + hh * 8 + k0;
      v16h af = frag_from_pair(*(const v8h*)(xp), *(const v8h*)(xp + 16));
      acc[mt] = __builtin_amdgcn_wmma_f32_16x16x32_f16(false, af, false, bf, (short)0,
                                                       acc[mt], false, false);
    }
  }
  int o = wave * 16 + mm;
  float bv = bias[(size_t)n * cout + o];
#pragma unroll
  for (int mt = 0; mt < 4; ++mt) {
#pragma unroll
    for (int r = 0; r < 8; ++r) {
      int b = mt * 16 + hh * 8 + r;
      float v = acc[mt][r] + bv;
      if (mode == 0) {
        float s = 1.f / (1.f + __expf(-v));
        size_t base = ((size_t)n * BATCH + b) * HIDC;
        if (o < HIDC) zbuf[base + o] = (f16)s;
        else          rbuf[base + (o - HIDC)] = s;
      } else {
        float hc = tanhf(v);
        size_t idx = ((size_t)n * BATCH + b) * HIDC + o;
        float rv = rbuf[idx];
        float hn = rv * h[idx] + (1.f - rv) * hc;
        h[idx] = hn;
        if (seq_out) seq_out[idx] = (f16)hn;
      }
    }
  }
}

// ---------------- final projection: out[b,hor,n,0] -------------------------
__global__ __launch_bounds__(256) void final_proj(const float* __restrict__ h,
                                                  const float* __restrict__ convw,
                                                  const float* __restrict__ convb,
                                                  float* __restrict__ out) {
  int idx = blockIdx.x * 256 + threadIdx.x;
  int total = BATCH * TSTEPS * NNODE;
  if (idx >= total) return;
  int n = idx % NNODE;
  int o = (idx / NNODE) % TSTEPS;
  int b = idx / (NNODE * TSTEPS);
  const float* hb = h + ((size_t)n * BATCH + b) * HIDC;
  const float* wo = convw + o * HIDC;
  float s = 0.f;
#pragma unroll
  for (int k = 0; k < HIDC; ++k) s += hb[k] * wo[k];
  out[idx] = s + convb[o];
}

// ---------------------------------------------------------------------------
static inline int ceil_div(int a, int b) { return (a + b - 1) / b; }

extern "C" void kernel_launch(void* const* d_in, const int* in_sizes, int n_in,
                              void* d_out, int out_size, void* d_ws, size_t ws_size,
                              hipStream_t stream) {
  (void)in_sizes; (void)n_in; (void)out_size; (void)ws_size;
  const float* source = (const float*)d_in[0];
  const float* E      = (const float*)d_in[1];
  const float* w0g    = (const float*)d_in[2];
  const float* b0g    = (const float*)d_in[3];
  const float* w0c    = (const float*)d_in[4];
  const float* b0c    = (const float*)d_in[5];
  const float* w1g    = (const float*)d_in[6];
  const float* b1g    = (const float*)d_in[7];
  const float* w1c    = (const float*)d_in[8];
  const float* b1c    = (const float*)d_in[9];
  const float* convw  = (const float*)d_in[10];
  const float* convb  = (const float*)d_in[11];
  float* out = (float*)d_out;

  char* ws = (char*)d_ws;
  size_t off = 0;
  auto carve = [&](size_t bytes) -> char* {
    off = (off + 255) & ~(size_t)255;
    char* p = ws + off;
    off += bytes;
    return p;
  };
  f16*   A   = (f16*)  carve((size_t)NPAD * NPAD * 2);
  f16*   W0G = (f16*)  carve((size_t)NNODE * 128 * K2P0 * 2);
  f16*   W0C = (f16*)  carve((size_t)NNODE * 64  * K2P0 * 2);
  f16*   W1G = (f16*)  carve((size_t)NNODE * 128 * K2P1 * 2);
  f16*   W1C = (f16*)  carve((size_t)NNODE * 64  * K2P1 * 2);
  float* B0G = (float*)carve((size_t)NNODE * 128 * 4);
  float* B0C = (float*)carve((size_t)NNODE * 64 * 4);
  float* B1G = (float*)carve((size_t)NNODE * 128 * 4);
  float* B1C = (float*)carve((size_t)NNODE * 64 * 4);
  float* H   = (float*)carve((size_t)NNODE * BATCH * HIDC * 4);
  float* Rb  = (float*)carve((size_t)NNODE * BATCH * HIDC * 4);
  f16*   Zb  = (f16*)  carve((size_t)NNODE * BATCH * HIDC * 2);
  f16*   Xn  = (f16*)  carve((size_t)NNODE * NC1 * 2);
  f16*   Xt  = (f16*)  carve((size_t)NC1 * NPAD * 2);
  f16*   AX  = (f16*)  carve((size_t)NPAD * NC1 * 2);
  f16*   SEQ = (f16*)  carve((size_t)TSTEPS * NNODE * BATCH * HIDC * 2);
  const size_t SEQT = (size_t)NNODE * BATCH * HIDC;

  // ---- one-time precompute (per call; deterministic) ----
  adj_kernel<<<NPAD, 256, 0, stream>>>(E, A);
  pool_wT<<<ceil_div(NNODE * 128 * K2P0, 256), 256, 0, stream>>>(W0G, w0g, E, CIN0, 128, K2P0);
  pool_wT<<<ceil_div(NNODE * 64  * K2P0, 256), 256, 0, stream>>>(W0C, w0c, E, CIN0, 64,  K2P0);
  pool_wT<<<ceil_div(NNODE * 128 * K2P1, 256), 256, 0, stream>>>(W1G, w1g, E, CIN1, 128, K2P1);
  pool_wT<<<ceil_div(NNODE * 64  * K2P1, 256), 256, 0, stream>>>(W1C, w1c, E, CIN1, 64,  K2P1);
  pool_b_kernel<<<ceil_div(NNODE * 128, 256), 256, 0, stream>>>(B0G, b0g, E, 128);
  pool_b_kernel<<<ceil_div(NNODE * 64, 256), 256, 0, stream>>>(B0C, b0c, E, 64);
  pool_b_kernel<<<ceil_div(NNODE * 128, 256), 256, 0, stream>>>(B1G, b1g, E, 128);
  pool_b_kernel<<<ceil_div(NNODE * 64, 256), 256, 0, stream>>>(B1C, b1c, E, 64);

  dim3 agg0(ceil_div(NC0 / 32, 8), NPAD / 32);
  dim3 agg1(ceil_div(NC1 / 32, 8), NPAD / 32);
  int bx0 = ceil_div(NC0 * NNODE, 256);
  int bx1 = ceil_div(NC1 * NNODE, 256);
  int zh = ceil_div(NNODE * BATCH * HIDC, 256);

  // ---- layer 0 ----
  zero_h_kernel<<<zh, 256, 0, stream>>>(H);
  for (int t = 0; t < TSTEPS; ++t) {
    build_x0<<<bx0, 256, 0, stream>>>(Xn, Xt, source, H, nullptr, t);
    gemm_agg<<<agg0, 256, 0, stream>>>(A, Xt, AX, NC0);
    node_gemm<<<NNODE, 256, 0, stream>>>(Xn, AX, W0G, B0G, CIN0, 128, K2P0, 0, Zb, Rb, H,
                                         nullptr);
    build_x0<<<bx0, 256, 0, stream>>>(Xn, Xt, source, H, Zb, t);
    gemm_agg<<<agg0, 256, 0, stream>>>(A, Xt, AX, NC0);
    node_gemm<<<NNODE, 256, 0, stream>>>(Xn, AX, W0C, B0C, CIN0, 64, K2P0, 1, Zb, Rb, H,
                                         SEQ + (size_t)t * SEQT);
  }

  // ---- layer 1 ----
  zero_h_kernel<<<zh, 256, 0, stream>>>(H);
  for (int t = 0; t < TSTEPS; ++t) {
    const f16* xt = SEQ + (size_t)t * SEQT;
    build_x1<<<bx1, 256, 0, stream>>>(Xn, Xt, xt, H, nullptr);
    gemm_agg<<<agg1, 256, 0, stream>>>(A, Xt, AX, NC1);
    node_gemm<<<NNODE, 256, 0, stream>>>(Xn, AX, W1G, B1G, CIN1, 128, K2P1, 0, Zb, Rb, H,
                                         nullptr);
    build_x1<<<bx1, 256, 0, stream>>>(Xn, Xt, xt, H, Zb);
    gemm_agg<<<agg1, 256, 0, stream>>>(A, Xt, AX, NC1);
    node_gemm<<<NNODE, 256, 0, stream>>>(Xn, AX, W1C, B1C, CIN1, 64, K2P1, 1, Zb, Rb, H,
                                         nullptr);
  }

  // ---- projection ----
  final_proj<<<ceil_div(BATCH * TSTEPS * NNODE, 256), 256, 0, stream>>>(H, convw, convb,
                                                                        out);
}
